// GCN_9345848836264
// MI455X (gfx1250) — matrix operand order repaired and checked by reading, not compiled, not run on hardware
//
#include <hip/hip_runtime.h>

typedef __attribute__((ext_vector_type(16))) _Float16 v16h;
typedef __attribute__((ext_vector_type(8)))  _Float16 v8h;
typedef __attribute__((ext_vector_type(8)))  float    v8f;

#define DIM_IN  256
#define DIM_HID 128
#define DIM_OUT 64

// ---------------- degree / normalization ----------------

__global__ void gcn_init_deg(float* __restrict__ deg, int n) {
  int i = blockIdx.x * blockDim.x + threadIdx.x;
  if (i < n) deg[i] = 1.0f;  // self-loop contributes 1
}

__global__ void gcn_deg_scatter(const int* __restrict__ dst, float* __restrict__ deg, int E) {
  int e = blockIdx.x * blockDim.x + threadIdx.x;
  if (e < E) atomicAdd(&deg[dst[e]], 1.0f);
}

__global__ void gcn_dinv(const float* __restrict__ deg, float* __restrict__ dinv, int n) {
  int i = blockIdx.x * blockDim.x + threadIdx.x;
  if (i < n) dinv[i] = rsqrtf(deg[i]);  // deg >= 1 always (self-loops)
}

// ---------------- weight preconversion: f32 KxN -> f16 NxK (transposed) ----------------

template<int K, int N>
__global__ void gcn_w_to_f16t(const float* __restrict__ W, _Float16* __restrict__ WT) {
  int idx = blockIdx.x * blockDim.x + threadIdx.x;  // over K*N, coalesced read
  if (idx >= K * N) return;
  int k = idx / N, n = idx % N;
  WT[(size_t)n * K + k] = (_Float16)W[idx];
}

// ---------------- dense GEMM via WMMA f16 -> f32 ----------------
// 256-thread block = 8 waves. Block stages its A rows (f32 -> f16) into padded
// LDS once; each wave computes one 16x16 tile. B fragments come from the
// pre-transposed f16 weight (contiguous 8-f16 reads, L0/L2-resident).

template<int K, int N>
__global__ void gcn_gemm_wmma(const float* __restrict__ A, const _Float16* __restrict__ WT,
                              float* __restrict__ D, int M) {
  constexpr int NT = N / 16;   // n-tiles covered by one block (8 waves)
  constexpr int MT = 8 / NT;   // m-tiles per block
  constexpr int R  = MT * 16;  // rows staged per block
  constexpr int LK = K + 8;    // padded LDS row stride (f16 elems; +16B kills bank conflicts)
  __shared__ _Float16 sA[R * LK];

  const int tid   = threadIdx.x;
  const int lane  = tid & 31;
  const int wave  = tid >> 5;
  const int mBase = blockIdx.x * R;

  // ---- cooperative A staging: coalesced float4 loads, f16 stores ----
  for (int i = tid * 4; i < R * K; i += 256 * 4) {
    int row = i / K, col = i % K;
    int grow = mBase + row;
    float4 v;
    if (grow < M) v = *(const float4*)(A + (size_t)grow * K + col);
    else          v = make_float4(0.f, 0.f, 0.f, 0.f);
    _Float16* p = &sA[row * LK + col];
    p[0] = (_Float16)v.x; p[1] = (_Float16)v.y;
    p[2] = (_Float16)v.z; p[3] = (_Float16)v.w;
  }
  __syncthreads();

  const int tm = blockIdx.x * MT + wave / NT;
  const int tn = wave % NT;
  if (tm * 16 >= M) return;
  const int half = lane >> 4;
  const int l16  = lane & 15;

  const _Float16* ar = &sA[((wave / NT) * 16 + l16) * LK];  // this lane's A row
  const _Float16* br = WT + (size_t)(tn * 16 + l16) * K;    // this lane's B column (contig K)

  v8f c = {};
#pragma unroll
  for (int kb = 0; kb < K; kb += 32) {
    // A 16x32 f16 layout: lane half h holds K {8h..8h+7} (v0..3) and {16+8h..23+8h} (v4..7)
    v8h a0 = *(const v8h*)(ar + kb + 8 * half);
    v8h a1 = *(const v8h*)(ar + kb + 16 + 8 * half);
    // B 32x16 f16 layout: lane half h holds K {8h..8h+7} (v0..3) and {16+8h..23+8h} (v4..7) for its column
    v8h b0 = *(const v8h*)(br + kb + 8 * half);
    v8h b1 = *(const v8h*)(br + kb + 16 + 8 * half);
    v16h a = __builtin_shufflevector(a0, a1, 0, 1, 2, 3, 4, 5, 6, 7, 8, 9, 10, 11, 12, 13, 14, 15);
    v16h b = __builtin_shufflevector(b0, b1, 0, 1, 2, 3, 4, 5, 6, 7, 8, 9, 10, 11, 12, 13, 14, 15);
    c = __builtin_amdgcn_wmma_f32_16x16x32_f16(
        /*neg_a=*/false, a, /*neg_b=*/false, b,
        /*c_mod=*/(short)0, c, /*reuse_a=*/false, /*reuse_b=*/false);
  }

  // D 16x16 f32: VGPR r holds row (r + 8*half), col = l16.
  float* drow = D + tn * 16 + l16;
#pragma unroll
  for (int r = 0; r < 8; ++r) {
    int grow = tm * 16 + r + 8 * half;
    if (grow < M) drow[(size_t)grow * N] = c[r];
  }
}

// Scalar tail for M not divisible by 16 (not hit for N_NODES=50000, kept for safety).
template<int K, int N>
__global__ void gcn_gemm_tail(const float* __restrict__ A, const float* __restrict__ W,
                              float* __restrict__ D, int mStart, int M) {
  int idx = blockIdx.x * blockDim.x + threadIdx.x;
  int rows = M - mStart;
  if (idx >= rows * N) return;
  int r = mStart + idx / N, col = idx % N;
  float acc = 0.f;
  for (int k = 0; k < K; ++k) acc += A[(size_t)r * K + k] * W[(size_t)k * N + col];
  D[(size_t)r * N + col] = acc;
}

// ---------------- aggregation ----------------

// Initialize aggregation with the self-loop term: out[i] = h[i] * dinv[i]^2
template<int F>
__global__ void gcn_selfloop_init(const float* __restrict__ h, const float* __restrict__ dinv,
                                  float* __restrict__ out, int n) {
  int idx = blockIdx.x * blockDim.x + threadIdx.x;
  if (idx >= n * F) return;
  float di = dinv[idx / F];
  out[idx] = h[idx] * di * di;
}

// One wave per edge: gather h[src] (float4 per lane), scale by dinv[src]*dinv[dst],
// scatter with f32 atomics into out[dst].
template<int F>
__global__ void gcn_edge_scatter(const int* __restrict__ src, const int* __restrict__ dst,
                                 const float* __restrict__ dinv,
                                 const float* __restrict__ h, float* __restrict__ out, int E) {
  const int lane = threadIdx.x & 31;
  int e = (blockIdx.x * blockDim.x + threadIdx.x) >> 5;
  if (e >= E) return;
  int s = src[e], d = dst[e];
  float norm = dinv[s] * dinv[d];
  constexpr int PL = F / 32;  // floats per lane (4 for 128 feats, 2 for 64)
  const float* hs = h + (size_t)s * F + lane * PL;
  float* od = out + (size_t)d * F + lane * PL;
  if constexpr (PL == 4) {
    float4 v = *(const float4*)hs;
    atomicAdd(od + 0, v.x * norm);
    atomicAdd(od + 1, v.y * norm);
    atomicAdd(od + 2, v.z * norm);
    atomicAdd(od + 3, v.w * norm);
  } else {
    float2 v = *(const float2*)hs;
    atomicAdd(od + 0, v.x * norm);
    atomicAdd(od + 1, v.y * norm);
  }
}

template<int F, bool RELU>
__global__ void gcn_bias_act(float* __restrict__ x, const float* __restrict__ b, int n) {
  int idx = blockIdx.x * blockDim.x + threadIdx.x;
  if (idx >= n * F) return;
  float v = x[idx] + b[idx & (F - 1)];
  x[idx] = RELU ? fmaxf(v, 0.0f) : v;
}

// ---------------- launch ----------------

extern "C" void kernel_launch(void* const* d_in, const int* in_sizes, int n_in,
                              void* d_out, int out_size, void* d_ws, size_t ws_size,
                              hipStream_t stream) {
  const float* x   = (const float*)d_in[0];
  const int*  edge = (const int*)d_in[1];
  const float* W1  = (const float*)d_in[2];
  const float* b1  = (const float*)d_in[3];
  const float* W2  = (const float*)d_in[4];
  const float* b2  = (const float*)d_in[5];

  const int n = in_sizes[0] / DIM_IN;   // 50000
  const int E = in_sizes[1] / 2;        // 800000
  const int* src = edge;
  const int* dst = edge + E;

  float* ws   = (float*)d_ws;
  float* deg  = ws;                                   // [n]
  float* dinv = ws + n;                               // [n]
  float* h    = ws + 2 * (size_t)n;                   // [n*128], reused as h2 [n*64]
  float* agg1 = h + (size_t)n * DIM_HID;              // [n*128]
  _Float16* wt1 = (_Float16*)(agg1 + (size_t)n * DIM_HID);  // [128*256] f16 (64KB)
  _Float16* wt2 = wt1 + DIM_IN * DIM_HID;                   // [64*128]  f16 (16KB)
  float* out  = (float*)d_out;                        // [n*64]

  const dim3 blk(256);
  auto grid1d = [](long long work) { return dim3((unsigned)((work + 255) / 256)); };

  // Degree + symmetric normalization
  gcn_init_deg<<<grid1d(n), blk, 0, stream>>>(deg, n);
  gcn_deg_scatter<<<grid1d(E), blk, 0, stream>>>(dst, deg, E);
  gcn_dinv<<<grid1d(n), blk, 0, stream>>>(deg, dinv, n);

  // Pre-convert weights: f32 KxN -> f16 NxK (done once per call; tiny)
  gcn_w_to_f16t<DIM_IN, DIM_HID><<<grid1d(DIM_IN * DIM_HID), blk, 0, stream>>>(W1, wt1);
  gcn_w_to_f16t<DIM_HID, DIM_OUT><<<grid1d(DIM_HID * DIM_OUT), blk, 0, stream>>>(W2, wt2);

  // Layer 1: h = x @ W1 (WMMA), aggregate, relu+bias
  {
    int mt = (n + 15) >> 4;
    int blocks = mt;  // MT = 1 for N=128
    gcn_gemm_wmma<DIM_IN, DIM_HID><<<dim3(blocks), blk, 0, stream>>>(x, wt1, h, n);
    if (n & 15)
      gcn_gemm_tail<DIM_IN, DIM_HID><<<grid1d((long long)(n & 15) * DIM_HID), blk, 0, stream>>>(
          x, W1, h, (n >> 4) << 4, n);
  }
  gcn_selfloop_init<DIM_HID><<<grid1d((long long)n * DIM_HID), blk, 0, stream>>>(h, dinv, agg1, n);
  gcn_edge_scatter<DIM_HID><<<grid1d((long long)E * 32), blk, 0, stream>>>(src, dst, dinv, h, agg1, E);
  gcn_bias_act<DIM_HID, true><<<grid1d((long long)n * DIM_HID), blk, 0, stream>>>(agg1, b1, n);

  // Layer 2: h2 = agg1 @ W2 (WMMA, reuse h buffer), aggregate into d_out, bias
  {
    int mt = (n + 15) >> 4;
    int blocks = (mt + 1) / 2;  // MT = 2 for N=64
    gcn_gemm_wmma<DIM_HID, DIM_OUT><<<dim3(blocks), blk, 0, stream>>>(agg1, wt2, h, n);
    if (n & 15)
      gcn_gemm_tail<DIM_HID, DIM_OUT><<<grid1d((long long)(n & 15) * DIM_OUT), blk, 0, stream>>>(
          agg1, W2, h, (n >> 4) << 4, n);
  }
  gcn_selfloop_init<DIM_OUT><<<grid1d((long long)n * DIM_OUT), blk, 0, stream>>>(h, dinv, out, n);
  gcn_edge_scatter<DIM_OUT><<<grid1d((long long)E * 32), blk, 0, stream>>>(src, dst, dinv, h, out, E);
  gcn_bias_act<DIM_OUT, false><<<grid1d((long long)n * DIM_OUT), blk, 0, stream>>>(out, b2, n);
}
